// GATv2GraphNet_ENZYMES_34832184770980
// MI455X (gfx1250) — matrix operand order repaired and checked by reading, not compiled, not run on hardware
//
#include <hip/hip_runtime.h>
#include <hip/hip_bf16.h>
#include <math.h>

typedef __attribute__((ext_vector_type(16))) _Float16 v16h;
typedef __attribute__((ext_vector_type(8)))  float    v8f;

#define NEG_SLOPE 0.2f

// ---------------------------------------------------------------------------
// WMMA GEMM: C[M,NOUT] = A[M,K] @ W[K,NOUT] (+ bias). f32 in/out, f16 MACs,
// f32 accumulate via v_wmma_f32_16x16x32_f16. One 16x16 tile per wave32.
// K, NOUT compile-time -> fully unrolled chain of K/32 WMMAs, 32-bit addressing.
// ---------------------------------------------------------------------------
template <int K, int NOUT>
__global__ void gat_wmma_gemm(const float* __restrict__ A,
                              const float* __restrict__ W,
                              const float* __restrict__ bias,
                              float* __restrict__ C,
                              int M) {
    const int lane   = threadIdx.x & 31;
    const int wave   = threadIdx.x >> 5;
    const int tilesN = NOUT >> 4;
    const int tilesM = (M + 15) >> 4;
    const int tile   = blockIdx.x * (blockDim.x >> 5) + wave;
    if (tile >= tilesM * tilesN) return;        // wave-uniform: EXEC stays all-1s

    const int tm     = (tile / tilesN) * 16;
    const int tn     = (tile % tilesN) * 16;
    const int laneHi = lane >> 4;               // 0: lanes 0-15, 1: lanes 16-31
    const int l15    = lane & 15;

    int row = tm + l15;                         // A row this lane feeds
    if (row >= M) row = M - 1;                  // clamp (keeps EXEC full)
    const int col = tn + l15;                   // C/B column this lane owns

    v8f acc = {0.f, 0.f, 0.f, 0.f, 0.f, 0.f, 0.f, 0.f};

    #pragma unroll
    for (int kk = 0; kk < K; kk += 32) {
        // A fragment (16x32 f16): elem e <-> K = kk + (e>>3)*16 + (e&7) + laneHi*8
        v16h a;
        const float4* ap = (const float4*)(A + row * K + kk + laneHi * 8);
        const float4 a0 = ap[0], a1 = ap[1];    // K+0..7
        const float4 a2 = ap[4], a3 = ap[5];    // K+16..23
        a[0]=(_Float16)a0.x; a[1]=(_Float16)a0.y; a[2]=(_Float16)a0.z; a[3]=(_Float16)a0.w;
        a[4]=(_Float16)a1.x; a[5]=(_Float16)a1.y; a[6]=(_Float16)a1.z; a[7]=(_Float16)a1.w;
        a[8]=(_Float16)a2.x; a[9]=(_Float16)a2.y; a[10]=(_Float16)a2.z; a[11]=(_Float16)a2.w;
        a[12]=(_Float16)a3.x; a[13]=(_Float16)a3.y; a[14]=(_Float16)a3.z; a[15]=(_Float16)a3.w;

        // B fragment (32x16 f16): elem e <-> K = kk + (e>>3)*16 + (e&7) + laneHi*8, col = lane&15
        v16h b;
        const float* wp = W + (kk + laneHi * 8) * NOUT + col;
        #pragma unroll
        for (int e = 0; e < 16; ++e)
            b[e] = (_Float16)wp[((e & 7) + (e >> 3) * 16) * NOUT];

        acc = __builtin_amdgcn_wmma_f32_16x16x32_f16(
                  false, a, false, b, (short)0, acc, false, false);
    }

    const float bv = bias ? bias[col] : 0.0f;
    // D layout: VGPR j -> M = tm + j + 8*laneHi, N = col
    if (tm + 16 <= M) {                         // full tile: unguarded stores
        float* cp = C + (tm + laneHi * 8) * NOUT + col;
        #pragma unroll
        for (int j = 0; j < 8; ++j)
            cp[j * NOUT] = acc[j] + bv;
    } else {
        #pragma unroll
        for (int j = 0; j < 8; ++j) {
            const int r = tm + laneHi * 8 + j;
            if (r < M) C[r * NOUT + col] = acc[j] + bv;
        }
    }
}

// ---------------------------------------------------------------------------
// helpers
// ---------------------------------------------------------------------------
__device__ __forceinline__ unsigned f32_ord_enc(float f) {
    unsigned u = __float_as_uint(f);
    return u ^ ((u & 0x80000000u) ? 0xFFFFFFFFu : 0x80000000u);
}
__device__ __forceinline__ float f32_ord_dec(unsigned u) {
    return __uint_as_float(u ^ ((u & 0x80000000u) ? 0x80000000u : 0xFFFFFFFFu));
}

__global__ void gat_zero_u32(unsigned* __restrict__ p, int n) {
    int i = blockIdx.x * blockDim.x + threadIdx.x;
    if (i < n) p[i] = 0u;
}

// ---------------------------------------------------------------------------
// Edge phase 1: raw attention score + segment max (ordered-uint atomicMax).
// Edges e < E come from edge_index; e >= E are self loops (node e-E). H=2.
// ---------------------------------------------------------------------------
template <int CC>
__global__ void gat_edge_score(const float* __restrict__ xl,
                               const float* __restrict__ xr,
                               const float* __restrict__ att,
                               const long long* __restrict__ ei,
                               int E, int Nn,
                               float* __restrict__ score,
                               unsigned* __restrict__ smax) {
    const int T = (E + Nn) * 2;
    int t = blockIdx.x * blockDim.x + threadIdx.x;
    if (t >= T) return;
    const int h = t & 1;
    const int e = t >> 1;
    const int src = (e < E) ? (int)ei[e]     : (e - E);
    const int dst = (e < E) ? (int)ei[E + e] : (e - E);

    const float* pl = xl + src * 2 * CC + h * CC;
    const float* pr = xr + dst * 2 * CC + h * CC;
    const float* pa = att + h * CC;
    float s = 0.0f;
    #pragma unroll
    for (int c = 0; c < CC; ++c) {
        float v = pl[c] + pr[c];
        v = (v > 0.0f) ? v : NEG_SLOPE * v;
        s += v * pa[c];
    }
    score[t] = s;
    atomicMax(&smax[dst * 2 + h], f32_ord_enc(s));
}

// Edge phase 2: exp(score - max) + segment sum (denominator).
__global__ void gat_edge_exp(const long long* __restrict__ ei,
                             int E, int Nn,
                             float* __restrict__ score,
                             const unsigned* __restrict__ smax,
                             float* __restrict__ denom) {
    const int T = (E + Nn) * 2;
    int t = blockIdx.x * blockDim.x + threadIdx.x;
    if (t >= T) return;
    const int h = t & 1;
    const int e = t >> 1;
    const int dst = (e < E) ? (int)ei[E + e] : (e - E);
    const float m = f32_ord_dec(smax[dst * 2 + h]);
    const float ex = __expf(score[t] - m);
    score[t] = ex;
    atomicAdd(&denom[dst * 2 + h], ex);
}

// Edge phase 3: alpha-weighted scatter of source messages into destination.
template <int CC>
__global__ void gat_edge_agg(const float* __restrict__ xl,
                             const long long* __restrict__ ei,
                             int E, int Nn,
                             const float* __restrict__ score,
                             const float* __restrict__ denom,
                             float* __restrict__ out) {
    const int T = (E + Nn) * 2;
    int t = blockIdx.x * blockDim.x + threadIdx.x;
    if (t >= T) return;
    const int h = t & 1;
    const int e = t >> 1;
    const int src = (e < E) ? (int)ei[e]     : (e - E);
    const int dst = (e < E) ? (int)ei[E + e] : (e - E);
    const float alpha = score[t] / denom[dst * 2 + h];
    const float* pl = xl + src * 2 * CC + h * CC;
    float* po = out + dst * 2 * CC + h * CC;
    #pragma unroll
    for (int c = 0; c < CC; ++c)
        atomicAdd(&po[c], alpha * pl[c]);
}

// out = elu(out + bias), in place. n = Nn*F, bias has F entries (F power of 2).
__global__ void gat_bias_elu(float* __restrict__ out,
                             const float* __restrict__ bias,
                             int n, int Fmask) {
    int i = blockIdx.x * blockDim.x + threadIdx.x;
    if (i >= n) return;
    float v = out[i] + bias[i & Fmask];
    out[i] = (v > 0.0f) ? v : (__expf(v) - 1.0f);
}

// Global mean-pool accumulation: per-node scatter into per-graph sums.
template <int F>
__global__ void gat_pool(const float* __restrict__ h,
                         const long long* __restrict__ batch,
                         int Nn,
                         float* __restrict__ sums,
                         float* __restrict__ cnt) {
    int i = blockIdx.x * blockDim.x + threadIdx.x;
    if (i >= Nn) return;
    const int g = (int)batch[i];
    const float* ph = h + i * F;
    float* ps = sums + g * F;
    #pragma unroll
    for (int c = 0; c < F; ++c) atomicAdd(&ps[c], ph[c]);
    atomicAdd(&cnt[g], 1.0f);
}

// mean -> fc(32x6) -> log_softmax, one thread per graph.
__global__ void gat_fc_logsoftmax(const float* __restrict__ sums,
                                  const float* __restrict__ cnt,
                                  const float* __restrict__ Wfc,
                                  const float* __restrict__ bfc,
                                  int G, int F, int NC,
                                  float* __restrict__ out) {
    int g = blockIdx.x * blockDim.x + threadIdx.x;
    if (g >= G) return;
    const float inv = 1.0f / fmaxf(cnt[g], 1.0f);
    float logit[8];
    for (int j = 0; j < NC; ++j) {
        float acc = bfc[j];
        for (int c = 0; c < F; ++c)
            acc += (sums[g * F + c] * inv) * Wfc[c * NC + j];
        logit[j] = acc;
    }
    float m = logit[0];
    for (int j = 1; j < NC; ++j) m = fmaxf(m, logit[j]);
    float s = 0.0f;
    for (int j = 0; j < NC; ++j) s += __expf(logit[j] - m);
    const float lse = m + __logf(s);
    for (int j = 0; j < NC; ++j) out[g * NC + j] = logit[j] - lse;
}

// ---------------------------------------------------------------------------
// host-side orchestration
// ---------------------------------------------------------------------------
static inline int cdiv_i(long long a, int b) { return (int)((a + b - 1) / b); }

extern "C" void kernel_launch(void* const* d_in, const int* in_sizes, int n_in,
                              void* d_out, int out_size, void* d_ws, size_t ws_size,
                              hipStream_t stream) {
    const float*     x     = (const float*)d_in[0];
    const long long* ei    = (const long long*)d_in[1];   // [2,E] int64
    const long long* batch = (const long long*)d_in[2];   // [N]  int64
    const float* Wl1 = (const float*)d_in[3];
    const float* bl1 = (const float*)d_in[4];
    const float* Wr1 = (const float*)d_in[5];
    const float* att1= (const float*)d_in[6];
    const float* b1  = (const float*)d_in[7];
    const float* Wl2 = (const float*)d_in[8];
    const float* bl2 = (const float*)d_in[9];
    const float* Wr2 = (const float*)d_in[10];
    const float* att2= (const float*)d_in[11];
    const float* b2  = (const float*)d_in[12];
    const float* Wfc = (const float*)d_in[13];
    const float* bfc = (const float*)d_in[14];

    const int Fin = 128, NC = 6, G = 512;
    const int N  = in_sizes[0] / Fin;
    const int E  = in_sizes[1] / 2;
    const int Et = E + N;                       // with self loops
    const int F1 = 64;                          // H*C1
    const int F2 = 32;                          // H*C2

    // workspace carve-out (256B aligned)
    char* ws = (char*)d_ws;
    auto carve = [&](size_t bytes) -> char* {
        char* p = ws; ws += (bytes + 255) & ~(size_t)255; return p;
    };
    float*    xl1   = (float*)carve((size_t)N * F1 * 4);
    float*    xr1   = (float*)carve((size_t)N * F1 * 4);
    float*    h1    = (float*)carve((size_t)N * F1 * 4);   // agg1 -> h1 in place
    float*    xl2   = (float*)carve((size_t)N * F2 * 4);
    float*    xr2   = (float*)carve((size_t)N * F2 * 4);
    float*    h2    = (float*)carve((size_t)N * F2 * 4);   // agg2 -> h2 in place
    float*    score = (float*)carve((size_t)Et * 2 * 4);
    unsigned* smax  = (unsigned*)carve((size_t)N * 2 * 4);
    float*    denom = (float*)carve((size_t)N * 2 * 4);
    float*    psum  = (float*)carve((size_t)G * F2 * 4);
    float*    pcnt  = (float*)carve((size_t)G * 4);
    (void)ws_size;

    const int BLK = 256;
    const int WPB = BLK / 32;                   // waves (tiles) per block
    const int TE  = Et * 2;

    const int tiles1 = ((N + 15) / 16) * (F1 / 16);
    const int tiles2 = ((N + 15) / 16) * (F2 / 16);
    const int gblk1  = (tiles1 + WPB - 1) / WPB;
    const int gblk2  = (tiles2 + WPB - 1) / WPB;

    // ---- layer 1: x[N,128] -> xl1/xr1[N,64] ----
    gat_wmma_gemm<128, 64><<<gblk1, BLK, 0, stream>>>(x, Wl1, bl1, xl1, N);
    gat_wmma_gemm<128, 64><<<gblk1, BLK, 0, stream>>>(x, Wr1, nullptr, xr1, N);
    gat_zero_u32<<<cdiv_i(N * 2, BLK), BLK, 0, stream>>>(smax, N * 2);
    gat_zero_u32<<<cdiv_i(N * 2, BLK), BLK, 0, stream>>>((unsigned*)denom, N * 2);
    gat_zero_u32<<<cdiv_i(N * F1, BLK), BLK, 0, stream>>>((unsigned*)h1, N * F1);
    gat_edge_score<32><<<cdiv_i(TE, BLK), BLK, 0, stream>>>(xl1, xr1, att1, ei, E, N, score, smax);
    gat_edge_exp      <<<cdiv_i(TE, BLK), BLK, 0, stream>>>(ei, E, N, score, smax, denom);
    gat_edge_agg<32>  <<<cdiv_i(TE, BLK), BLK, 0, stream>>>(xl1, ei, E, N, score, denom, h1);
    gat_bias_elu      <<<cdiv_i(N * F1, BLK), BLK, 0, stream>>>(h1, b1, N * F1, F1 - 1);

    // ---- layer 2: h1[N,64] -> xl2/xr2[N,32] ----
    gat_wmma_gemm<64, 32><<<gblk2, BLK, 0, stream>>>(h1, Wl2, bl2, xl2, N);
    gat_wmma_gemm<64, 32><<<gblk2, BLK, 0, stream>>>(h1, Wr2, nullptr, xr2, N);
    gat_zero_u32<<<cdiv_i(N * 2, BLK), BLK, 0, stream>>>(smax, N * 2);
    gat_zero_u32<<<cdiv_i(N * 2, BLK), BLK, 0, stream>>>((unsigned*)denom, N * 2);
    gat_zero_u32<<<cdiv_i(N * F2, BLK), BLK, 0, stream>>>((unsigned*)h2, N * F2);
    gat_edge_score<16><<<cdiv_i(TE, BLK), BLK, 0, stream>>>(xl2, xr2, att2, ei, E, N, score, smax);
    gat_edge_exp      <<<cdiv_i(TE, BLK), BLK, 0, stream>>>(ei, E, N, score, smax, denom);
    gat_edge_agg<16>  <<<cdiv_i(TE, BLK), BLK, 0, stream>>>(xl2, ei, E, N, score, denom, h2);
    gat_bias_elu      <<<cdiv_i(N * F2, BLK), BLK, 0, stream>>>(h2, b2, N * F2, F2 - 1);

    // ---- pool + fc + log_softmax ----
    gat_zero_u32<<<cdiv_i(G * F2, BLK), BLK, 0, stream>>>((unsigned*)psum, G * F2);
    gat_zero_u32<<<cdiv_i(G, BLK), BLK, 0, stream>>>((unsigned*)pcnt, G);
    gat_pool<32><<<cdiv_i(N, BLK), BLK, 0, stream>>>(h2, batch, N, psum, pcnt);
    gat_fc_logsoftmax<<<cdiv_i(G, BLK), BLK, 0, stream>>>(psum, pcnt, Wfc, bfc, G, F2, NC, (float*)d_out);
}
